// PointInstanceLoss_52673478918522
// MI455X (gfx1250) — compile-verified, matrix operand order checked by reading, not run on hardware
//
#include <hip/hip_runtime.h>
#include <hip/hip_bf16.h>

#define KINST 16
#define DV 0.5f
#define DD 1.5f
#define GAMMAC 0.001f
#define CTILES 8   // column tiles per wave in the boundary GEMM

typedef __attribute__((ext_vector_type(16))) _Float16 v16h;
typedef __attribute__((ext_vector_type(8)))  _Float16 v8h;
typedef __attribute__((ext_vector_type(8)))  float    v8f;

__device__ inline float fast_sqrtf(float x) { return __builtin_amdgcn_sqrtf(x); }

// ---------------------------------------------------------------- k0: init
__global__ void pil_init(float* out, float* cnt, float* fcsum, float* gcsum, int B) {
  int t = blockIdx.x * blockDim.x + threadIdx.x;
  int stride = gridDim.x * blockDim.x;
  if (t < 7) out[t] = 0.f;
  for (int i = t; i < B * KINST; i += stride) cnt[i] = 0.f;
  for (int i = t; i < B * KINST * 64; i += stride) fcsum[i] = 0.f;
  for (int i = t; i < B * KINST * 3; i += stride) gcsum[i] = 0.f;
}

// ------------------------------------------- k1: normalize + accumulate sums
// one wave32 per point; lane holds 2 of the 64 dims
__global__ void pil_normalize(const float* __restrict__ emb,
                              const float* __restrict__ pts,
                              const int*   __restrict__ lab,
                              float* emb_n, _Float16* emb_h,
                              float* cnt, float* fcsum, float* gcsum, int N) {
  int p    = (blockIdx.x * blockDim.x + threadIdx.x) >> 5;   // point id in [0,B*N)
  int lane = threadIdx.x & 31;
  const float2 v = ((const float2*)(emb + (size_t)p * 64))[lane];
  float ss = v.x * v.x + v.y * v.y;
  #pragma unroll
  for (int off = 16; off > 0; off >>= 1) ss += __shfl_xor(ss, off, 32);
  float sc = 1.f / fmaxf(fast_sqrtf(ss), 1e-12f);
  float e0 = v.x * sc, e1 = v.y * sc;
  size_t base = (size_t)p * 64 + lane * 2;
  emb_n[base] = e0;            emb_n[base + 1] = e1;
  emb_h[base] = (_Float16)e0;  emb_h[base + 1] = (_Float16)e1;
  int L = lab[p];
  if (L >= 1 && L <= KINST) {
    int slot = (p / N) * KINST + (L - 1);
    if (lane == 0) atomicAdd(&cnt[slot], 1.f);
    atomicAdd(&fcsum[slot * 64 + lane * 2 + 0], e0);
    atomicAdd(&fcsum[slot * 64 + lane * 2 + 1], e1);
    if (lane < 3) atomicAdd(&gcsum[slot * 3 + lane], pts[(size_t)p * 3 + lane]);
  }
}

// ------------------------------------------------------------- k2: centers
__global__ void pil_centers(const float* cnt, const float* fcsum, const float* gcsum,
                            float* fc, float* cen, float* gc, float* fcn, int B) {
  int t = threadIdx.x;
  if (t >= B * KINST) return;
  float c = fmaxf(cnt[t], 1.f);           // every id present by construction
  float n2 = 0.f;
  for (int d = 0; d < 64; ++d) {
    float f = fcsum[t * 64 + d] / c;
    fc[t * 64 + d] = f; n2 += f * f;
  }
  float nrm = fast_sqrtf(n2);
  fcn[t] = nrm;
  float isc = 1.f / fmaxf(nrm, 1e-12f);
  for (int d = 0; d < 64; ++d) cen[t * 64 + d] = fc[t * 64 + d] * isc;
  for (int j = 0; j < 3; ++j) gc[t * 3 + j] = gcsum[t * 3 + j] / c;
}

// --------------------------------------------- k3: dist_loss + reg_loss (per b)
__global__ void pil_distreg(const float* cen, float* out, float invBe) {
  __shared__ float sc[KINST * 64];
  __shared__ float red[256];
  int b = blockIdx.x, t = threadIdx.x;
  for (int i = t; i < KINST * 64; i += 256) sc[i] = cen[(size_t)b * KINST * 64 + i];
  __syncthreads();
  float part = 0.f;
  int k1 = t >> 4, k2 = t & 15;
  if (k1 != k2) {
    float d2 = 0.f;
    for (int d = 0; d < 64; ++d) { float df = sc[k1*64+d] - sc[k2*64+d]; d2 += df*df; }
    part = fmaxf(2.f * DD - fast_sqrtf(fmaxf(d2, 0.f)), 0.f);
  }
  red[t] = part; __syncthreads();
  for (int s = 128; s > 0; s >>= 1) { if (t < s) red[t] += red[t + s]; __syncthreads(); }
  if (t == 0) atomicAdd(&out[2], red[0] * (1.f / (KINST * (KINST - 1))) * invBe);
  __syncthreads();
  float r = 0.f;
  for (int i = t; i < KINST * 64; i += 256) r += sc[i] * sc[i];
  red[t] = r; __syncthreads();
  for (int s = 128; s > 0; s >>= 1) { if (t < s) red[t] += red[t + s]; __syncthreads(); }
  if (t == 0) atomicAdd(&out[3], fast_sqrtf(red[0]) * GAMMAC * invBe);
}

// ----------------------- k4: per-point var/center terms + sim[B,K,N] for lovasz
__global__ void pil_point(const float* __restrict__ emb_n,
                          const float* __restrict__ pts,
                          const int*   __restrict__ lab,
                          const float* fc, const float* gc, const float* cnt,
                          const float* fcn, float* sim_t,
                          float* out, int N, float invBe, float invB) {
  __shared__ float sfc[KINST * 64];
  __shared__ float sgc[KINST * 3];
  __shared__ float scnt[KINST], sfcn[KINST];
  __shared__ float red[256];
  int t = threadIdx.x;
  int p = blockIdx.x * 256 + t;
  int b = p / N, n = p % N;
  for (int i = t; i < KINST * 64; i += 256) sfc[i] = fc[(size_t)b * KINST * 64 + i];
  if (t < KINST * 3) sgc[t] = gc[b * KINST * 3 + t];
  if (t < KINST) { scnt[t] = cnt[b * KINST + t]; sfcn[t] = fcn[b * KINST + t]; }
  __syncthreads();
  float4 er[16];
  const float4* rp = (const float4*)(emb_n + (size_t)p * 64);
  #pragma unroll
  for (int i = 0; i < 16; ++i) er[i] = rp[i];
  float s[KINST];
  #pragma unroll
  for (int k = 0; k < KINST; ++k) {
    const float4* fp = (const float4*)(sfc + k * 64);
    float acc = 0.f;
    #pragma unroll
    for (int i = 0; i < 16; ++i) {
      float4 f = fp[i];
      acc += er[i].x*f.x + er[i].y*f.y + er[i].z*f.z + er[i].w*f.w;
    }
    s[k] = acc;
    sim_t[((size_t)(b * KINST + k)) * N + n] = acc / fmaxf(sfcn[k], 1e-8f);
  }
  float varp = 0.f, cenp = 0.f;
  int L = lab[p];
  if (L >= 1 && L <= KINST) {
    int l = L - 1;
    float fn = sfcn[l];
    float scen = s[l] / fmaxf(fn, 1e-12f);
    float cen2 = (fn > 0.f) ? 1.f : 0.f;
    float dnc = fast_sqrtf(fmaxf(1.f + cen2 - 2.f * scen, 0.f));   // |emb_n| == 1
    varp = fmaxf(dnc - DV, 0.f) / scnt[l];
    float fd = fast_sqrtf(fmaxf(1.f + fn * fn - 2.f * s[l], 0.f));
    float dx = pts[(size_t)p*3+0] - sgc[l*3+0];
    float dy = pts[(size_t)p*3+1] - sgc[l*3+1];
    float dz = pts[(size_t)p*3+2] - sgc[l*3+2];
    cenp = fast_sqrtf(dx*dx + dy*dy + dz*dz) * fd / scnt[l];
  }
  red[t] = varp; __syncthreads();
  for (int s2 = 128; s2 > 0; s2 >>= 1) { if (t < s2) red[t] += red[t + s2]; __syncthreads(); }
  if (t == 0) atomicAdd(&out[1], red[0] * invBe);
  __syncthreads();
  red[t] = cenp; __syncthreads();
  for (int s2 = 128; s2 > 0; s2 >>= 1) { if (t < s2) red[t] += red[t + s2]; __syncthreads(); }
  if (t == 0) atomicAdd(&out[4], red[0] * invB);
}

// ------------------- k5: boundary loss via WMMA f32<=f16 16x16x32 Gram tiles
// 16-bit A layout (ISA 7.12.2): lane L holds row M=L%16; lanes 0-15 hold
// K = {kb..kb+7, kb+16..kb+23}, lanes 16-31 hold K = {kb+8..kb+15, kb+24..kb+31}.
// B = E^T, so B's columns are E's rows -> same gather for both operands.
__device__ inline v16h pil_load_tile(const _Float16* Eb, int row0, int kb, int lane) {
  const _Float16* r = Eb + (size_t)(row0 + (lane & 15)) * 64;
  int hi = (lane >> 4) & 1;
  v8h c0 = *(const v8h*)(r + kb + hi * 8);
  v8h c1 = *(const v8h*)(r + kb + 16 + hi * 8);
  v16h v;
  #pragma unroll
  for (int j = 0; j < 8; ++j) { v[j] = c0[j]; v[8 + j] = c1[j]; }
  return v;
}

// one wave computes a 16-row strip x (CTILES*16) columns: A loaded once,
// reused across CTILES column tiles -> 2*CTILES static v_wmma per wave.
__global__ void pil_boundary(const _Float16* __restrict__ emb_h,
                             const int* __restrict__ lab,
                             float* out, int N, int TM, float bscale) {
  int wave = (blockIdx.x * blockDim.x + threadIdx.x) >> 5;   // exact grid, no guard
  int lane = threadIdx.x & 31;
  int TC   = TM / CTILES;              // column groups per row strip
  int perB = TM * TC;
  int b  = wave / perB;
  int r  = wave % perB;
  int i0 = (r / TC) * 16;
  int j0 = (r % TC) * (16 * CTILES);
  const _Float16* Eb = emb_h + (size_t)b * N * 64;
  const int* labB = lab + b * N;

  v16h a0 = pil_load_tile(Eb, i0, 0,  lane);
  v16h a1 = pil_load_tile(Eb, i0, 32, lane);

  // D layout: lanes 0-15 -> N=lane, M=vgpr; lanes 16-31 -> N=lane-16, M=8+vgpr
  int n = lane & 15;
  int mbase = (lane >> 4) * 8;
  int lrow[8];
  #pragma unroll
  for (int rr = 0; rr < 8; ++rr) lrow[rr] = labB[i0 + mbase + rr];

  float part = 0.f;
  #pragma unroll
  for (int ct = 0; ct < CTILES; ++ct) {
    int jc = j0 + ct * 16;
    v16h b0 = pil_load_tile(Eb, jc, 0,  lane);
    v16h b1 = pil_load_tile(Eb, jc, 32, lane);
    v8f acc = {};
    acc = __builtin_amdgcn_wmma_f32_16x16x32_f16(
        false, a0, false, b0, (short)0, acc, false, false);
    acc = __builtin_amdgcn_wmma_f32_16x16x32_f16(
        false, a1, false, b1, (short)0, acc, false, false);
    int lj = labB[jc + n];
    #pragma unroll
    for (int rr = 0; rr < 8; ++rr) {
      float s = acc[rr];
      // branchless: same-instance pairs contribute the full hinge DD (bd==0)
      float dm = fast_sqrtf(fmaxf(2.f - 2.f * s, 0.f));
      float hinge = fmaxf(DD - dm, 0.f);
      part += (lrow[rr] == lj) ? DD : hinge;
    }
  }
  __shared__ float red[256];
  red[threadIdx.x] = part; __syncthreads();
  for (int s2 = 128; s2 > 0; s2 >>= 1) {
    if (threadIdx.x < s2) red[threadIdx.x] += red[threadIdx.x + s2];
    __syncthreads();
  }
  if (threadIdx.x == 0) atomicAdd(&out[5], red[0] * bscale);
}

// -------------------------- k6: lovasz, one block per (b,k), bitonic sort N=4096
__global__ void pil_lovasz(const float* __restrict__ sim_t,
                           const int* __restrict__ lab,
                           const float* cnt, float* out, int N, float invB) {
  __shared__ float key[4096];
  __shared__ float pay[4096];
  __shared__ float bsum[512];
  int bk = blockIdx.x;                       // b*K + k
  int b = bk / KINST, k = bk % KINST;
  int t = threadIdx.x;
  const float* S  = sim_t + (size_t)bk * N;
  const int*   Lb = lab + b * N;
  for (int i = t; i < 4096; i += 512) {
    key[i] = S[i];
    pay[i] = (Lb[i] == k + 1) ? 1.f : 0.f;
  }
  float cntk = cnt[bk];
  // descending bitonic sort
  for (int size = 2; size <= 4096; size <<= 1)
    for (int stride = size >> 1; stride > 0; stride >>= 1) {
      __syncthreads();
      for (int q = t; q < 2048; q += 512) {
        int lo = 2 * q - (q & (stride - 1));
        int hi = lo + stride;
        bool desc = ((lo & size) == 0);
        float ka = key[lo], kb = key[hi];
        if (desc ? (ka < kb) : (ka > kb)) {
          key[lo] = kb; key[hi] = ka;
          float pa = pay[lo]; pay[lo] = pay[hi]; pay[hi] = pa;
        }
      }
    }
  __syncthreads();
  // blocked inclusive scan of pay (8 elements / thread)
  int base = t * 8;
  float chunk = 0.f;
  #pragma unroll
  for (int j = 0; j < 8; ++j) chunk += pay[base + j];
  bsum[t] = chunk;
  __syncthreads();
  for (int off = 1; off < 512; off <<= 1) {
    float v = (t >= off) ? bsum[t - off] : 0.f;
    __syncthreads();
    bsum[t] += v;
    __syncthreads();
  }
  float run = bsum[t] - chunk;               // exclusive prefix
  #pragma unroll
  for (int j = 0; j < 8; ++j) { run += pay[base + j]; pay[base + j] = run; }
  __syncthreads();
  float part = 0.f;
  #pragma unroll
  for (int j = 0; j < 8; ++j) {
    int idx = base + j;
    float c  = pay[idx];
    float cp = (idx == 0) ? 0.f : pay[idx - 1];
    float jac  = 1.f - (cntk - c)  / (cntk + (float)(idx + 1) - c);
    float jacp = (idx == 0) ? 0.f
               : 1.f - (cntk - cp) / (cntk + (float)idx - cp);
    part += fmaxf(key[idx], 0.f) * (jac - jacp);
  }
  __syncthreads();
  bsum[t] = part; __syncthreads();
  for (int s2 = 256; s2 > 0; s2 >>= 1) { if (t < s2) bsum[t] += bsum[t + s2]; __syncthreads(); }
  if (t == 0) atomicAdd(&out[6], bsum[0] * invB);
}

// --------------------------------------------------------------- k7: total
__global__ void pil_total(float* out) {
  if (threadIdx.x == 0 && blockIdx.x == 0) {
    out[0] = 0.1f * (out[1] + out[2] + out[3])
           + 0.1f * out[4] + 0.05f * out[5] + 0.01f * out[6];
  }
}

extern "C" void kernel_launch(void* const* d_in, const int* in_sizes, int n_in,
                              void* d_out, int out_size, void* d_ws, size_t ws_size,
                              hipStream_t stream) {
  const float* pts = (const float*)d_in[0];
  const float* emb = (const float*)d_in[1];
  const int*   lab = (const int*)d_in[2];
  float* out = (float*)d_out;

  const int B = 4;
  const int N = in_sizes[2] / B;             // 4096
  const float invBe = 1.f / ((float)B + 1e-6f);
  const float invB  = 1.f / (float)B;

  // workspace carve-out (256B aligned slices)
  char* w = (char*)d_ws;
  auto carve = [&](size_t bytes) -> void* {
    void* p = (void*)w;
    w += (bytes + 255) & ~(size_t)255;
    return p;
  };
  float*     emb_n = (float*)    carve((size_t)B * N * 64 * sizeof(float));
  _Float16*  emb_h = (_Float16*) carve((size_t)B * N * 64 * sizeof(_Float16));
  float*     sim_t = (float*)    carve((size_t)B * KINST * N * sizeof(float));
  float*     cnt   = (float*)    carve((size_t)B * KINST * sizeof(float));
  float*     fcsum = (float*)    carve((size_t)B * KINST * 64 * sizeof(float));
  float*     fc    = (float*)    carve((size_t)B * KINST * 64 * sizeof(float));
  float*     cen   = (float*)    carve((size_t)B * KINST * 64 * sizeof(float));
  float*     gcsum = (float*)    carve((size_t)B * KINST * 3 * sizeof(float));
  float*     gc    = (float*)    carve((size_t)B * KINST * 3 * sizeof(float));
  float*     fcn   = (float*)    carve((size_t)B * KINST * sizeof(float));

  pil_init<<<4, 256, 0, stream>>>(out, cnt, fcsum, gcsum, B);
  pil_normalize<<<(B * N) / 8, 256, 0, stream>>>(emb, pts, lab, emb_n, emb_h,
                                                 cnt, fcsum, gcsum, N);
  pil_centers<<<1, 64, 0, stream>>>(cnt, fcsum, gcsum, fc, cen, gc, fcn, B);
  pil_distreg<<<B, 256, 0, stream>>>(cen, out, invBe);
  pil_point<<<(B * N) / 256, 256, 0, stream>>>(emb_n, pts, lab, fc, gc, cnt, fcn,
                                               sim_t, out, N, invBe, invB);
  const int TM = N / 16;                     // 256 row strips per batch
  const int wavesPerB = TM * (TM / CTILES);  // 256 * 32
  const int totalWaves = B * wavesPerB;      // 32768, divisible by 8
  const float bscale = invB / ((float)N * (float)N);
  pil_boundary<<<totalWaves / 8, 256, 0, stream>>>(emb_h, lab, out, N, TM, bscale);
  pil_lovasz<<<B * KINST, 512, 0, stream>>>(sim_t, lab, cnt, out, N, invB);
  pil_total<<<1, 64, 0, stream>>>(out);
}
